// GraphSAGE_10866267259418
// MI455X (gfx1250) — matrix operand order repaired and checked by reading, not compiled, not run on hardware
//
#include <hip/hip_runtime.h>

// ---------------------------------------------------------------------------
// GraphSAGE (2-layer, mean aggregation) for MI455X / gfx1250, wave32 + WMMA.
//  - GEMMs via v_wmma_f32_16x16x32_bf16; 80-row blocks amortize weight staging
//  - weights pre-converted to bf16 once (no per-block f32 reads / cvt)
//  - scatter-mean via global f32 atomics (the true bandwidth bottleneck)
//  - layer-2 aggregation AFTER the neighbor GEMM (mean is linear) -> 128-wide
// ---------------------------------------------------------------------------

typedef __attribute__((ext_vector_type(16))) __bf16 v16bf;
typedef __attribute__((ext_vector_type(8)))  float  v8f;

#define DEVINL __device__ __forceinline__

DEVINL unsigned short f2bf(float f) {
  unsigned int u = __builtin_bit_cast(unsigned int, f);
  unsigned int r = (u + 0x7FFFu + ((u >> 16) & 1u)) >> 16;  // RNE
  return (unsigned short)r;
}

// ------------------------------- utility kernels ---------------------------

__global__ __launch_bounds__(256) void zero_f32(float* p, int n) {
  for (int i = blockIdx.x * blockDim.x + threadIdx.x; i < n;
       i += gridDim.x * blockDim.x)
    p[i] = 0.0f;
}

__global__ __launch_bounds__(256) void cvt_f32_bf16(const float* __restrict__ in,
                                                    unsigned short* __restrict__ out,
                                                    int n) {
  for (int i = blockIdx.x * blockDim.x + threadIdx.x; i < n;
       i += gridDim.x * blockDim.x)
    out[i] = f2bf(in[i]);
}

__global__ __launch_bounds__(256) void count_deg(const int* __restrict__ dst,
                                                 float* __restrict__ cnt, int E) {
  int e = blockIdx.x * blockDim.x + threadIdx.x;
  if (e < E) atomicAdd(&cnt[dst[e]], 1.0f);
}

__global__ __launch_bounds__(256) void inv_deg(float* __restrict__ d, int n) {
  int i = blockIdx.x * blockDim.x + threadIdx.x;
  if (i < n) d[i] = 1.0f / fmaxf(d[i], 1.0f);
}

// one edge per 32 lanes; each lane moves 4 floats (128-wide rows)
__global__ __launch_bounds__(256) void scatter_add(const int* __restrict__ src,
                                                   const int* __restrict__ dst,
                                                   const float* __restrict__ val,
                                                   float* __restrict__ out, int E) {
  int gid = blockIdx.x * blockDim.x + threadIdx.x;
  int e = gid >> 5;
  if (e >= E) return;
  int l = gid & 31;
  const float4 v = *(const float4*)(val + (size_t)src[e] * 128 + l * 4);
  float* o = out + (size_t)dst[e] * 128 + l * 4;
  atomicAdd(o + 0, v.x);
  atomicAdd(o + 1, v.y);
  atomicAdd(o + 2, v.z);
  atomicAdd(o + 3, v.w);
}

__global__ __launch_bounds__(256) void scale_rows(float* __restrict__ u,
                                                  const float* __restrict__ s, int nrow) {
  int gid = blockIdx.x * blockDim.x + threadIdx.x;
  int row = gid >> 5;
  if (row >= nrow) return;
  int l = gid & 31;
  float sc = s[row];
  float4* p = (float4*)(u + (size_t)row * 128 + l * 4);
  float4 v = *p;
  v.x *= sc; v.y *= sc; v.z *= sc; v.w *= sc;
  *p = v;
}

// ------------------------------- WMMA GEMM ---------------------------------
// C[M x NOUT] = A[M x K] (f32 or bf16) @ Wb[K x NOUT] (bf16) + bias.
// One block = 80 output rows (M = 80*1250 exactly); 8 waves each own one
// 16-col tile and carry 5 accumulators (one per 16-row sub-tile), reusing the
// B operand registers across 5 consecutive WMMAs.
// KPAD = K + 8 -> 16B-aligned, bank-conflict-free ds_load_b128 fetches.

template <int K, int NOUT, bool ABF16>
__global__ __launch_bounds__(256) void gemm_wmma(const void* __restrict__ A_,
                                                 const unsigned short* __restrict__ Wb,
                                                 const float* __restrict__ bias,
                                                 float* __restrict__ Out) {
  constexpr int MROWS = 80;
  constexpr int RT = MROWS / 16;            // 5 row sub-tiles
  constexpr int KPAD = K + 8;
  constexpr int NPAD = (NOUT + 15) & ~15;
  constexpr int NT = NPAD / 16;
  __shared__ unsigned short As[MROWS * KPAD];
  __shared__ unsigned short Wt[NPAD * KPAD];

  const int tid = threadIdx.x;
  const int row0 = blockIdx.x * MROWS;

  // Stage A strip [80 x K] -> bf16 LDS; 4-wide global loads, 8B LDS stores.
  if constexpr (ABF16) {
    const unsigned short* A = (const unsigned short*)A_ + (size_t)row0 * K;
    for (int idx = tid; idx < MROWS * K / 4; idx += 256) {
      int m = idx / (K / 4), k4 = idx % (K / 4);
      *(ushort4*)(As + m * KPAD + k4 * 4) = ((const ushort4*)A)[idx];
    }
  } else {
    const float* A = (const float*)A_ + (size_t)row0 * K;
    for (int idx = tid; idx < MROWS * K / 4; idx += 256) {
      int m = idx / (K / 4), k4 = idx % (K / 4);
      const float4 v = ((const float4*)A)[idx];
      ushort4 o; o.x = f2bf(v.x); o.y = f2bf(v.y); o.z = f2bf(v.z); o.w = f2bf(v.w);
      *(ushort4*)(As + m * KPAD + k4 * 4) = o;
    }
  }
  // Stage Wb [K x NOUT] (already bf16) transposed -> Wt[n][k].
  if constexpr ((NOUT & 15) != 0) {
    for (int idx = tid; idx < NPAD * KPAD; idx += 256) Wt[idx] = 0;
    __syncthreads();
  }
  for (int idx = tid; idx < K * NOUT; idx += 256) {
    int k = idx / NOUT, n = idx % NOUT;
    Wt[n * KPAD + k] = Wb[idx];
  }
  __syncthreads();

  const int wv = tid >> 5;
  if (wv < NT) {
    const int lane = tid & 31;
    const int half = lane >> 4;   // lanes 16-31 hold the upper K/M halves
    const int mn = lane & 15;     // row for A, col for B/C

    union AB { v16bf v; uint4 q[2]; };
    AB a, b;
    v8f c[RT];
#pragma unroll
    for (int r = 0; r < RT; ++r) c[r] = (v8f){0.f,0.f,0.f,0.f,0.f,0.f,0.f,0.f};

    const unsigned short* arow = As + mn * KPAD;
    const unsigned short* brow = Wt + (wv * 16 + mn) * KPAD;
#pragma unroll
    for (int kk = 0; kk < K; kk += 32) {
      // B 32x16 bf16: lanes 0-15 k=0..15, lanes 16-31 k=16..31 (Wt is [n][k]).
      b.q[0] = *(const uint4*)(brow + kk + half * 16);
      b.q[1] = *(const uint4*)(brow + kk + half * 16 + 8);
#pragma unroll
      for (int r = 0; r < RT; ++r) {
        // A 16x32 bf16: lanes 0-15 k=0..15, lanes 16-31 k=16..31 (matches B).
        const unsigned short* ar = arow + r * 16 * KPAD;
        a.q[0] = *(const uint4*)(ar + kk + half * 16);
        a.q[1] = *(const uint4*)(ar + kk + half * 16 + 8);
        c[r] = __builtin_amdgcn_wmma_f32_16x16x32_bf16(false, a.v, false, b.v,
                                                       (short)0, c[r], false, false);
      }
    }

    const int col = wv * 16 + mn;
    if ((NOUT & 15) == 0 || col < NOUT) {
      float bv = bias ? bias[col] : 0.0f;
#pragma unroll
      for (int r = 0; r < RT; ++r) {
#pragma unroll
        for (int rr = 0; rr < 8; ++rr) {  // C layout: M = rr + 8*half, N = mn
          Out[(size_t)(row0 + r * 16 + half * 8 + rr) * NOUT + col] = c[r][rr] + bv;
        }
      }
    }
  }
}

// --------------------- concat + ReLU + L2-normalize ------------------------
// One row per block (256 threads == 256 output features). Neighbor half can
// be scaled by 1/deg and biased here (layer 2 folds those in post-scatter).

__global__ __launch_bounds__(256) void concat_relu_norm(
    const float* __restrict__ Sx, const float* __restrict__ Ngh,
    const float* __restrict__ invdeg, const float* __restrict__ bias_n,
    unsigned short* __restrict__ H) {
  const int row = blockIdx.x;
  const int col = threadIdx.x;  // 0..255

  float v;
  if (col < 128) {
    v = Sx[(size_t)row * 128 + col];
  } else {
    v = Ngh[(size_t)row * 128 + (col - 128)];
    if (invdeg) v *= invdeg[row];
    if (bias_n) v += bias_n[col - 128];
  }
  v = fmaxf(v, 0.0f);

  float ss = v * v;
#pragma unroll
  for (int off = 16; off > 0; off >>= 1) ss += __shfl_xor(ss, off, 32);
  __shared__ float red[8];
  if ((threadIdx.x & 31) == 0) red[threadIdx.x >> 5] = ss;
  __syncthreads();
  float tot = 0.0f;
#pragma unroll
  for (int i = 0; i < 8; ++i) tot += red[i];

  float denom = fmaxf(sqrtf(tot), 1e-12f);
  H[(size_t)row * 256 + col] = f2bf(v / denom);
}

// ------------------------------- host driver -------------------------------

extern "C" void kernel_launch(void* const* d_in, const int* in_sizes, int n_in,
                              void* d_out, int out_size, void* d_ws, size_t ws_size,
                              hipStream_t stream) {
  (void)in_sizes; (void)n_in; (void)out_size; (void)ws_size;
  const int N = 100000;
  const int E = 1600000;

  const float* x   = (const float*)d_in[0];
  const int*   src = (const int*)d_in[1];
  const int*   dst = (const int*)d_in[2];
  const float* w1s = (const float*)d_in[3];
  const float* b1s = (const float*)d_in[4];
  const float* w1n = (const float*)d_in[5];
  const float* b1n = (const float*)d_in[6];
  const float* w2s = (const float*)d_in[7];
  const float* b2s = (const float*)d_in[8];
  const float* w2n = (const float*)d_in[9];
  const float* b2n = (const float*)d_in[10];
  const float* wfc = (const float*)d_in[11];
  const float* bfc = (const float*)d_in[12];
  float* out = (float*)d_out;

  char* ws = (char*)d_ws;
  size_t p = 0;
  auto take = [&](size_t bytes) -> char* {
    char* q = ws + p;
    p += (bytes + 255) & ~(size_t)255;
    return q;
  };
  float* invdeg       = (float*)take((size_t)N * 4);           // degree -> 1/deg
  float* S            = (float*)take((size_t)N * 128 * 4);     // self-branch GEMM out
  float* T            = (float*)take((size_t)N * 128 * 4);     // neigh-branch GEMM out
  float* U            = (float*)take((size_t)N * 128 * 4);     // scatter accumulator
  unsigned short* H1  = (unsigned short*)take((size_t)N * 256 * 2);  // bf16 acts
  unsigned short* H2  = (unsigned short*)take((size_t)N * 256 * 2);  // bf16 acts
  unsigned short* Wb1s = (unsigned short*)take((size_t)128 * 128 * 2);
  unsigned short* Wb1n = (unsigned short*)take((size_t)128 * 128 * 2);
  unsigned short* Wb2s = (unsigned short*)take((size_t)256 * 128 * 2);
  unsigned short* Wb2n = (unsigned short*)take((size_t)256 * 128 * 2);
  unsigned short* Wbfc = (unsigned short*)take((size_t)256 * 40 * 2);

  const int gemmBlocks = N / 80;  // 1250 (exact)
  const int eThreads = E * 32;    // one edge per 32 lanes

  // ---- one-time bf16 weight conversion (tiny) ----
  cvt_f32_bf16<<<64, 256, 0, stream>>>(w1s, Wb1s, 128 * 128);
  cvt_f32_bf16<<<64, 256, 0, stream>>>(w1n, Wb1n, 128 * 128);
  cvt_f32_bf16<<<128, 256, 0, stream>>>(w2s, Wb2s, 256 * 128);
  cvt_f32_bf16<<<128, 256, 0, stream>>>(w2n, Wb2n, 256 * 128);
  cvt_f32_bf16<<<40, 256, 0, stream>>>(wfc, Wbfc, 256 * 40);

  // ---- degrees -> 1/deg ----
  zero_f32<<<1024, 256, 0, stream>>>(invdeg, N);
  count_deg<<<(E + 255) / 256, 256, 0, stream>>>(dst, invdeg, E);
  inv_deg<<<(N + 255) / 256, 256, 0, stream>>>(invdeg, N);

  // ---- layer 1: neigh = mean_agg(x); h1 = l2norm(relu([x@w1s+b1s | neigh@w1n+b1n]))
  zero_f32<<<8192, 256, 0, stream>>>(U, N * 128);
  scatter_add<<<(eThreads + 255) / 256, 256, 0, stream>>>(src, dst, x, U, E);
  scale_rows<<<(N * 32 + 255) / 256, 256, 0, stream>>>(U, invdeg, N);
  gemm_wmma<128, 128, false><<<gemmBlocks, 256, 0, stream>>>(x, Wb1s, b1s, S);
  gemm_wmma<128, 128, false><<<gemmBlocks, 256, 0, stream>>>(U, Wb1n, b1n, T);
  concat_relu_norm<<<N, 256, 0, stream>>>(S, T, nullptr, nullptr, H1);

  // ---- layer 2: GEMM-first aggregation (mean is linear): z = h1@w2n, then
  //      scatter-mean z (128-wide instead of 256-wide), bias folded in norm.
  gemm_wmma<256, 128, true><<<gemmBlocks, 256, 0, stream>>>(H1, Wb2s, b2s, S);
  gemm_wmma<256, 128, true><<<gemmBlocks, 256, 0, stream>>>(H1, Wb2n, nullptr, T);
  zero_f32<<<8192, 256, 0, stream>>>(U, N * 128);
  scatter_add<<<(eThreads + 255) / 256, 256, 0, stream>>>(src, dst, T, U, E);
  concat_relu_norm<<<N, 256, 0, stream>>>(S, U, invdeg, b2n, H2);

  // ---- head: out = h2 @ wfc + bfc  (NOUT=40, column-masked stores)
  gemm_wmma<256, 40, true><<<gemmBlocks, 256, 0, stream>>>(H2, Wbfc, bfc, out);
}
